// GeometricStructureEmbedding_56831007261093
// MI455X (gfx1250) — matrix operand order repaired and checked by reading, compiled, not run on hardware
//
#include <hip/hip_runtime.h>
#include <hip/hip_bf16.h>
#include <math.h>

#define N_PTS 384
#define HID 256
#define HALF_H 128

typedef __attribute__((ext_vector_type(16))) _Float16 v16h;
typedef __attribute__((ext_vector_type(8)))  float    v8f;

union F16x16 { v16h v; _Float16 e[16]; };
union F32x8  { v8f  v; float    e[8];  };

// ---------------- prep: convert weights to f16 ----------------
__global__ void gse_cvt_kernel(const float* __restrict__ Wd, const float* __restrict__ Wa,
                               _Float16* __restrict__ Wd_h, _Float16* __restrict__ Wa_h) {
    int i = blockIdx.x * blockDim.x + threadIdx.x;
    if (i < HID * HID) {
        Wd_h[i] = (_Float16)Wd[i];
        Wa_h[i] = (_Float16)Wa[i];
    }
}

// ---------------- prep: top-3 nearest neighbors -> reference vectors ----------------
__global__ void gse_nn_kernel(const float* __restrict__ points, float* __restrict__ refvec) {
    int n = blockIdx.x * blockDim.x + threadIdx.x;
    if (n >= N_PTS) return;
    const float pnx = points[n*3+0], pny = points[n*3+1], pnz = points[n*3+2];
    float bd0 = 1e30f, bd1 = 1e30f, bd2 = 1e30f;
    int   bi0 = 0,     bi1 = 0,     bi2 = 0;
    for (int m = 0; m < N_PTS; ++m) {
        if (m == n) continue;
        const float dx = points[m*3+0] - pnx;
        const float dy = points[m*3+1] - pny;
        const float dz = points[m*3+2] - pnz;
        const float d2 = dx*dx + dy*dy + dz*dz;
        if (d2 < bd0)      { bd2=bd1; bi2=bi1; bd1=bd0; bi1=bi0; bd0=d2; bi0=m; }
        else if (d2 < bd1) { bd2=bd1; bi2=bi1; bd1=d2;  bi1=m; }
        else if (d2 < bd2) { bd2=d2;  bi2=m; }
    }
    const int bi[3] = { bi0, bi1, bi2 };
    #pragma unroll
    for (int k = 0; k < 3; ++k) {
        refvec[n*9 + k*3 + 0] = points[bi[k]*3+0] - pnx;
        refvec[n*9 + k*3 + 1] = points[bi[k]*3+1] - pny;
        refvec[n*9 + k*3 + 2] = points[bi[k]*3+2] - pnz;
    }
}

// ---------------- fragment loaders ----------------
// A-matrix 16x32 f16 layout (ISA 7.12.2): lanes 0-15: row M=lane, K pairs
// {0..7} in v0-3 and {16..23} in v4-7; lanes 16-31: same row, K {8..15},{24..31}.
__device__ __forceinline__ v16h load_fragA(const _Float16* __restrict__ base, int lane, int kbase) {
    const int r  = lane & 15;
    const int hs = (lane >> 4) << 3;          // 0 or 8
    const _Float16* row = base + r * HID;
    F16x16 u;
    #pragma unroll
    for (int v = 0; v < 4; ++v) {
        u.e[2*v]     = row[kbase + hs + 2*v];
        u.e[2*v + 1] = row[kbase + hs + 2*v + 1];
    }
    #pragma unroll
    for (int v = 0; v < 4; ++v) {
        u.e[8 + 2*v]     = row[kbase + 16 + hs + 2*v];
        u.e[8 + 2*v + 1] = row[kbase + 16 + hs + 2*v + 1];
    }
    return u.v;
}

// B-matrix 32x16 f16 layout: lane holds column N=lane&15; lanes 0-15 hold
// K=kbase+{0..7,16..23}, lanes 16-31 hold K=kbase+{8..15,24..31}.
// B[k,n] = W[n,k]  (out = E * W^T), so B column n == W row (obase+n).
__device__ __forceinline__ v16h load_fragB(const _Float16* __restrict__ W, int obase, int lane, int kbase) {
    const int col  = lane & 15;
    const int koff = kbase + ((lane >> 4) << 3);  // +0 or +8
    const _Float16* row = W + (size_t)(obase + col) * HID + koff;
    F16x16 u;
    #pragma unroll
    for (int i = 0; i < 16; ++i) u.e[i] = row[(i & 7) + ((i >> 3) << 4)];
    return u.v;
}

// ---------------- fused main kernel ----------------
__global__ __launch_bounds__(256) void gse_main_kernel(
    const float* __restrict__ points,
    const _Float16* __restrict__ Wd_h,
    const _Float16* __restrict__ Wa_h,
    const float* __restrict__ refvec,
    float* __restrict__ out)
{
    __shared__ _Float16 sEd[16 * HID];       //  8 KB: dist encoding, 16 rows x 256
    __shared__ _Float16 sEa[3 * 16 * HID];   // 24 KB: angle encodings, k x 16 rows x 256

    const int tid     = threadIdx.x;
    const int nm_base = blockIdx.x * 16;

    // ---- Phase 1: encodings into LDS (16 threads per row, 8 freqs each) ----
    {
        const int r  = tid >> 4;
        const int f0 = tid & 15;
        const int nm = nm_base + r;
        const int n  = nm / N_PTS;
        const int m  = nm - n * N_PTS;

        const float pnx = points[n*3+0], pny = points[n*3+1], pnz = points[n*3+2];
        // anc = diffs[n,m] = p_n - p_m
        const float ax = pnx - points[m*3+0];
        const float ay = pny - points[m*3+1];
        const float az = pnz - points[m*3+2];
        const float dist = sqrtf(ax*ax + ay*ay + az*az);

        float xs[4];
        xs[0] = dist * 5.0f;                              // dist / SIGMA_D
        #pragma unroll
        for (int k = 0; k < 3; ++k) {
            const float rx = refvec[n*9 + k*3 + 0];
            const float ry = refvec[n*9 + k*3 + 1];
            const float rz = refvec[n*9 + k*3 + 2];
            const float cx = ry*az - rz*ay;
            const float cy = rz*ax - rx*az;
            const float cz = rx*ay - ry*ax;
            const float sv = sqrtf(cx*cx + cy*cy + cz*cz);
            const float cv = rx*ax + ry*ay + rz*az;
            xs[1+k] = atan2f(sv, cv) * 3.8197186342054885f;   // * FACTOR_A
        }

        #pragma unroll
        for (int j = 0; j < 8; ++j) {
            const int h = f0 + 16*j;
            const float freq = exp2f((float)h * -0.1038102529652301f); // -log2(1e4)/128
            #pragma unroll
            for (int s = 0; s < 4; ++s) {
                const float arg = xs[s] * freq;
                _Float16* dst = (s == 0) ? sEd : (sEa + (s-1) * 16 * HID);
                dst[r*HID + h]          = (_Float16)__sinf(arg);
                dst[r*HID + h + HALF_H] = (_Float16)__cosf(arg);
            }
        }
    }
    __syncthreads();

    // ---- Phase 2: WMMA GEMM, 8 waves x 2 output-col tiles each ----
    const int lane = tid & 31;
    const int wave = tid >> 5;

    #pragma unroll
    for (int t = 0; t < 2; ++t) {
        const int obase = (wave + 8*t) * 16;
        v8f accD = {}; v8f acc0 = {}; v8f acc1 = {}; v8f acc2 = {};

        for (int ks = 0; ks < 8; ++ks) {
            const int kbase = ks * 32;
            const v16h aD = load_fragA(sEd,               lane, kbase);
            const v16h a0 = load_fragA(sEa,               lane, kbase);
            const v16h a1 = load_fragA(sEa + 16*HID,      lane, kbase);
            const v16h a2 = load_fragA(sEa + 32*HID,      lane, kbase);
            const v16h bD = load_fragB(Wd_h, obase, lane, kbase);
            const v16h bA = load_fragB(Wa_h, obase, lane, kbase);
            accD = __builtin_amdgcn_wmma_f32_16x16x32_f16(false, aD, false, bD, (short)0, accD, false, false);
            acc0 = __builtin_amdgcn_wmma_f32_16x16x32_f16(false, a0, false, bA, (short)0, acc0, false, false);
            acc1 = __builtin_amdgcn_wmma_f32_16x16x32_f16(false, a1, false, bA, (short)0, acc1, false, false);
            acc2 = __builtin_amdgcn_wmma_f32_16x16x32_f16(false, a2, false, bA, (short)0, acc2, false, false);
        }

        F32x8 d, x0, x1, x2;
        d.v = accD; x0.v = acc0; x1.v = acc1; x2.v = acc2;
        const int Nc = lane & 15;
        const int Mh = (lane >> 4) * 8;
        #pragma unroll
        for (int j = 0; j < 8; ++j) {
            const float amax = fmaxf(fmaxf(x0.e[j], x1.e[j]), x2.e[j]);
            out[(size_t)(nm_base + Mh + j) * HID + obase + Nc] = d.e[j] + amax;
        }
    }
}

// ---------------- launch ----------------
extern "C" void kernel_launch(void* const* d_in, const int* in_sizes, int n_in,
                              void* d_out, int out_size, void* d_ws, size_t ws_size,
                              hipStream_t stream) {
    const float* points = (const float*)d_in[0];
    const float* Wd     = (const float*)d_in[1];
    const float* Wa     = (const float*)d_in[2];
    float* out          = (float*)d_out;

    _Float16* Wd_h  = (_Float16*)d_ws;                 // 128 KB
    _Float16* Wa_h  = Wd_h + HID * HID;                // 128 KB
    float*    refv  = (float*)(Wa_h + HID * HID);      // 384*9*4 B

    gse_cvt_kernel<<<(HID*HID + 255) / 256, 256, 0, stream>>>(Wd, Wa, Wd_h, Wa_h);
    gse_nn_kernel<<<(N_PTS + 127) / 128, 128, 0, stream>>>(points, refv);
    gse_main_kernel<<<(N_PTS * N_PTS) / 16, 256, 0, stream>>>(points, Wd_h, Wa_h, refv, out);
}
